// net_crossInteraction_66468913873310
// MI455X (gfx1250) — matrix operand order repaired and checked by reading, not compiled, not run on hardware
//
#include <hip/hip_runtime.h>

typedef __attribute__((ext_vector_type(16))) __bf16 v16bf;
typedef __attribute__((ext_vector_type(8)))  __bf16 v8bf;   // one 16-byte load
typedef __attribute__((ext_vector_type(8)))  float  v8f;

#define LRELU(x) ((x) > 0.f ? (x) : 0.1f * (x))

static __device__ __forceinline__ float block_reduce_sum(float v, float* sm) {
    int t = threadIdx.x;
    sm[t] = v; __syncthreads();
    for (int s = blockDim.x >> 1; s > 0; s >>= 1) {
        if (t < s) sm[t] += sm[t + s];
        __syncthreads();
    }
    float r = sm[0]; __syncthreads();
    return r;
}

// ---------------- utility kernels ----------------
__global__ void k_zero(float* p, int n) {
    int i = blockIdx.x * blockDim.x + threadIdx.x;
    if (i < n) p[i] = 0.f;
}

__global__ void k_f2bf(const float* __restrict__ in, __bf16* __restrict__ out, int n) {
    int i = blockIdx.x * blockDim.x + threadIdx.x;
    if (i < n) out[i] = (__bf16)in[i];
}

// convert with output row stride (K-padded destination)
__global__ void k_f2bf_pad(const float* __restrict__ in, __bf16* __restrict__ out,
                           int rows, int cols, int ldo) {
    int i = blockIdx.x * blockDim.x + threadIdx.x;
    if (i < rows * cols) {
        int r = i / cols, c = i - r * cols;
        out[(size_t)r * ldo + c] = (__bf16)in[i];
    }
}

// Bt[n*K+k] = W[k*N+n]  (store B transposed, bf16)
__global__ void k_transconv(const float* __restrict__ W, __bf16* __restrict__ Bt, int K, int N) {
    int i = blockIdx.x * blockDim.x + threadIdx.x;
    if (i < K * N) {
        int n = i / K, k = i - n * K;
        Bt[(size_t)n * K + k] = (__bf16)W[(size_t)k * N + n];
    }
}

// ---------------- WMMA bf16 GEMM:  C[M,N] = A[M,K] @ Bt[N,K]^T ----------------
// 128 threads = 4 waves; each wave computes a 16x64 C strip (4 wmma per K-step).
// Steady state is pure global_load_b128 + v_wmma: operand buffers are carved with
// row slack so no load guards are needed; only K must be a multiple of 32 (buffers
// K-padded with zeros where required). Store bounds (M,N) are enforced in epilogue.
#define NT 4
__global__ void wmma_gemm_bt(const __bf16* __restrict__ A, const __bf16* __restrict__ Bt,
                             float* __restrict__ C, __bf16* __restrict__ aux,
                             const float* __restrict__ bias,
                             int M, int N, int K, int lda, int ldb, int ldc,
                             long long sA, long long sB, long long sC, int aux_relu)
{
    int lane = threadIdx.x & 31;
    int wid  = threadIdx.x >> 5;
    int hf   = lane >> 4;      // wave half
    int l16  = lane & 15;
    int m0 = (blockIdx.y * 4 + wid) * 16;
    if (m0 >= ((M + 15) & ~15)) return;        // wave-uniform exit
    int n0 = blockIdx.x * (16 * NT);
    int b  = blockIdx.z;

    const v8bf* Ar = (const v8bf*)(A + (size_t)b * sA + (size_t)(m0 + l16) * lda);
    const v8bf* Br0 = (const v8bf*)(Bt + (size_t)b * sB + (size_t)(n0 + l16) * ldb);
    size_t bstep = ((size_t)16 * ldb) >> 3;    // 16 rows, in v8bf units

    v8f acc[NT] = {};
    int ksteps = K >> 5;
    for (int ks = 0; ks < ksteps; ++ks) {
        int k8 = ks * 4;
        union { v16bf v; v8bf h[2]; } ua, ub;
        // A 16x32 fragment: lane half hf -> chunks at K = k0+hf*8 and k0+16+hf*8
        ua.h[0] = Ar[k8 + hf];
        ua.h[1] = Ar[k8 + 2 + hf];
        #pragma unroll
        for (int t = 0; t < NT; ++t) {
            // B 32x16 fragment: lane = column, chunks at K = k0+hf*8 and k0+16+hf*8 (same as A)
            const v8bf* Br = Br0 + (size_t)t * bstep;
            ub.h[0] = Br[k8 + hf];
            ub.h[1] = Br[k8 + 2 + hf];
            acc[t] = __builtin_amdgcn_wmma_f32_16x16x32_bf16(false, ua.v, false, ub.v,
                                                             (short)0, acc[t], false, false);
        }
    }

    // C/D layout: lane -> n = lane&15 ; vgpr v -> m = v + 8*hf
    #pragma unroll
    for (int t = 0; t < NT; ++t) {
        int n = n0 + t * 16 + l16;
        if (n >= N) continue;
        float bz = bias ? bias[n] : 0.f;
        #pragma unroll
        for (int v = 0; v < 8; ++v) {
            int m = m0 + v + 8 * hf;
            if (m < M) {
                float r = acc[t][v] + bz;
                size_t o = (size_t)b * sC + (size_t)m * ldc + n;
                if (C)   C[o] = r;
                if (aux) aux[o] = (__bf16)(aux_relu ? fmaxf(r, 0.f) : r);
            }
        }
    }
}

// ---------------- small dense layers (GCN etc.), Nout == 256 ----------------
__global__ void k_linear(const float* __restrict__ X, const float* __restrict__ W,
                         const float* __restrict__ bias, float* __restrict__ Y,
                         __bf16* __restrict__ aux, int Kin, int relu_in)
{
    int r = blockIdx.x, n = threadIdx.x;
    const float* x = X + (size_t)r * Kin;
    float acc = bias[n];
    for (int k = 0; k < Kin; ++k) {
        float xv = x[k];
        if (relu_in) xv = fmaxf(xv, 0.f);
        acc = fmaf(xv, W[(size_t)k * 256 + n], acc);
    }
    Y[(size_t)r * 256 + n] = acc;
    if (aux) aux[(size_t)r * 256 + n] = (__bf16)acc;
}

// Z[b,i,:] = relu(adj[b,i,:] @ Y[b,:,:])
__global__ void k_adjrelu(const float* __restrict__ adj, const float* __restrict__ Yin,
                          float* __restrict__ Z)
{
    int i = blockIdx.x, b = blockIdx.y, d = threadIdx.x;
    const float* ar = adj + ((size_t)b * 56 + i) * 56;
    const float* yb = Yin + (size_t)b * 56 * 256;
    float acc = 0.f;
    #pragma unroll
    for (int j = 0; j < 56; ++j) acc = fmaf(ar[j], yb[j * 256 + d], acc);
    Z[((size_t)b * 56 + i) * 256 + d] = fmaxf(acc, 0.f);
}

// ---------------- sigmoid + per-batch sum ----------------
__global__ void k_sigsum(float* __restrict__ inter, float* __restrict__ sums)
{
    __shared__ float sm[256];
    int b = blockIdx.y;
    int idx = blockIdx.x * 256 + threadIdx.x;
    float v = 0.f;
    if (idx < 56000) {
        float L = inter[(size_t)b * 56000 + idx];
        v = 1.f / (1.f + expf(-L));
        inter[(size_t)b * 56000 + idx] = v;
    }
    float tot = block_reduce_sum(v, sm);
    if (threadIdx.x == 0) atomicAdd(&sums[b], tot);
}

// normalize in place; emit interT bf16 [b,k,i] (K-padded to 1024, pre-zeroed);
// loss1 + l1 partials
__global__ void k_norm(float* __restrict__ inter, const float* __restrict__ sums,
                       __bf16* __restrict__ interT, const float* __restrict__ prot_inter,
                       const float* __restrict__ exist,
                       float* __restrict__ l1a, float* __restrict__ d2a)
{
    __shared__ float sm[256];
    int b = blockIdx.y;
    int idx = blockIdx.x * 256 + threadIdx.x;
    float l1p = 0.f, d2p = 0.f;
    if (idx < 56000) {
        size_t o = (size_t)b * 56000 + idx;
        float v = inter[o] / sums[b];
        inter[o] = v;
        int i = idx / 56, k = idx - 56 * i;
        interT[(size_t)b * 65536 + (size_t)k * 1024 + i] = (__bf16)v;
        float lab = exist[b] * prot_inter[o];
        float df = v - lab;
        d2p = df * df;
        l1p = fabsf(v);
    }
    float t1 = block_reduce_sum(l1p, sm);
    if (threadIdx.x == 0) atomicAdd(&l1a[b], t1);
    float t2 = block_reduce_sum(d2p, sm);
    if (threadIdx.x == 0) atomicAdd(&d2a[b], t2);
}

// s[b,i] = sum_k inter[b,i,k]^2  (atomic-free row reduction)
__global__ void k_rowsq(const float* __restrict__ inter, float* __restrict__ sbuf)
{
    int idx = blockIdx.x * blockDim.x + threadIdx.x;   // 8000 rows
    if (idx < 8000) {
        const float* r = inter + (size_t)idx * 56;
        float s = 0.f;
        #pragma unroll 8
        for (int k = 0; k < 56; ++k) s = fmaf(r[k], r[k], s);
        sbuf[idx] = s;
    }
}

// fused tanh reduction into 64 contention slices:
// cp_part[i&63][b][d] += sum_k tanh(prot[b,i,d]*comp[b,k,d]) * inter[b,i,k]
__global__ void k_cp(const float* __restrict__ prot, const float* __restrict__ comp,
                     const float* __restrict__ inter, float* __restrict__ cp_part)
{
    int i = blockIdx.x, b = blockIdx.y, d = threadIdx.x;
    float p = prot[((size_t)b * 1000 + i) * 256 + d];
    const float* cm = comp + (size_t)b * 56 * 256;
    const float* w  = inter + ((size_t)b * 1000 + i) * 56;
    float acc = 0.f;
    #pragma unroll 4
    for (int k = 0; k < 56; ++k)
        acc = fmaf(tanhf(p * cm[k * 256 + d]), w[k], acc);
    atomicAdd(&cp_part[(size_t)(i & 63) * 2048 + b * 256 + d], acc);
}

// cp[b,d] = sum over 64 slices
__global__ void k_cpreduce(const float* __restrict__ cp_part, float* __restrict__ cp)
{
    int j = blockIdx.x * 256 + threadIdx.x;            // 2048
    float s = 0.f;
    #pragma unroll 8
    for (int sl = 0; sl < 64; ++sl) s += cp_part[(size_t)sl * 2048 + j];
    cp[j] = s;
}

// group reg: one block per (b,k): g = sum_i s[b,i]*contacts[b,k,i]; rs = row-sum
__global__ void k_group(const float* __restrict__ contacts, const float* __restrict__ sbuf,
                        float* __restrict__ reggroup)
{
    __shared__ float sm[256];
    int kk = blockIdx.x, b = blockIdx.y, t = threadIdx.x;
    const float* cr = contacts + ((size_t)b * 1000 + kk) * 1000;
    const float* sb = sbuf + b * 1000;
    float g = 0.f, rs = 0.f;
    for (int i = t; i < 1000; i += 256) {
        float c = cr[i];
        g = fmaf(sb[i], c, g);
        rs += c;
    }
    float gt = block_reduce_sum(g, sm);
    float rt = block_reduce_sum(rs, sm);
    if (t == 0) {
        gt = (gt == 0.f) ? 1e10f : gt;
        atomicAdd(&reggroup[b], sqrtf(gt) * sqrtf(rt));
    }
}

__global__ void k_fusedabs(const float* __restrict__ fo, float* __restrict__ facc)
{
    __shared__ float sm[256];
    int b = blockIdx.y;
    int idx = blockIdx.x * 256 + threadIdx.x;
    float v = (idx < 55944) ? fabsf(fo[(size_t)b * 55944 + idx]) : 0.f;
    float t = block_reduce_sum(v, sm);
    if (threadIdx.x == 0) atomicAdd(&facc[b], t);
}

// conv1d(stride2,pad1,k4) + leaky + maxpool4 -> h2[b, c*32+p]
__global__ void k_convpool(const float* __restrict__ cp, const float* __restrict__ cw,
                           const float* __restrict__ cb, float* __restrict__ h2)
{
    int b = blockIdx.x;
    const float* in = cp + (size_t)b * 256;
    for (int t = threadIdx.x; t < 2048; t += 256) {
        int c = t >> 5, p = t & 31;
        float mx = -3.4e38f;
        #pragma unroll
        for (int o4 = 0; o4 < 4; ++o4) {
            int o = p * 4 + o4;
            float a = cb[c];
            #pragma unroll
            for (int tt = 0; tt < 4; ++tt) {
                int j = 2 * o + tt - 1;
                if (j >= 0 && j < 256) a = fmaf(cw[c * 4 + tt], in[j], a);
            }
            a = LRELU(a);
            mx = fmaxf(mx, a);
        }
        h2[(size_t)b * 2048 + t] = mx;
    }
}

__global__ void k_mlp(const float* __restrict__ X, const float* __restrict__ W,
                      const float* __restrict__ bias, float* __restrict__ Y,
                      int Kin, int No)
{
    int b = blockIdx.y;
    int o = blockIdx.x * 256 + threadIdx.x;
    if (o >= No) return;
    const float* x = X + (size_t)b * Kin;
    float a = bias[o];
    for (int k = 0; k < Kin; ++k) a = fmaf(x[k], W[(size_t)k * No + o], a);
    Y[(size_t)b * No + o] = LRELU(a);
}

__global__ void k_final(const float* l1a, const float* facc, const float* rg,
                        const float* d2a, const float* r2, const float* Wr3,
                        const float* br3, const float* label, float* out)
{
    float ll1 = 0.f, lf = 0.f, lg = 0.f, l1s = 0.f, l2 = 0.f;
    for (int b = 0; b < 8; ++b) {
        ll1 += l1a[b]; lf += facc[b]; lg += rg[b]; l1s += sqrtf(d2a[b]);
        float a = br3[0];
        for (int k = 0; k < 300; ++k) a = fmaf(r2[b * 300 + k], Wr3[k], a);
        float df = a - label[b];
        l2 += df * df;
    }
    out[0] = (ll1 + lf + lg) * 0.125f + l1s * 0.125f + l2 * 0.125f;
}

// ---------------- host side ----------------
extern "C" void kernel_launch(void* const* d_in, const int* in_sizes, int n_in,
                              void* d_out, int out_size, void* d_ws, size_t ws_size,
                              hipStream_t stream)
{
    (void)in_sizes; (void)n_in; (void)out_size; (void)ws_size;
    const float* prot_data = (const float*)d_in[0];
    const float* drug_ver  = (const float*)d_in[1];
    const float* drug_adj  = (const float*)d_in[2];
    const float* contacts  = (const float*)d_in[3];
    const float* prot_int  = (const float*)d_in[4];
    const float* exist     = (const float*)d_in[5];
    const float* label     = (const float*)d_in[6];
    const float* fused     = (const float*)d_in[7];
    const float* W_prot = (const float*)d_in[8];  const float* b_prot = (const float*)d_in[9];
    const float* gW0 = (const float*)d_in[10];    const float* gb0 = (const float*)d_in[11];
    const float* gW1 = (const float*)d_in[12];    const float* gb1 = (const float*)d_in[13];
    const float* gW2 = (const float*)d_in[14];    const float* gb2 = (const float*)d_in[15];
    const float* gWf = (const float*)d_in[16];    const float* gbf = (const float*)d_in[17];
    const float* Wjp = (const float*)d_in[18];    const float* bjp = (const float*)d_in[19];
    const float* Wjc = (const float*)d_in[20];    const float* bjc = (const float*)d_in[21];
    const float* convw = (const float*)d_in[22];  const float* convb = (const float*)d_in[23];
    const float* Wr1 = (const float*)d_in[24];    const float* br1 = (const float*)d_in[25];
    const float* Wr2 = (const float*)d_in[26];    const float* br2 = (const float*)d_in[27];
    const float* Wr3 = (const float*)d_in[28];    const float* br3 = (const float*)d_in[29];

    char* w = (char*)d_ws;
    size_t off = 0;
    auto carve = [&](size_t bytes) -> void* {
        void* p = w + off;
        off = (off + bytes + 255) & ~(size_t)255;
        return p;
    };
    // WMMA operand buffers carved with row slack so fragment loads need no guards.
    __bf16* A1       = (__bf16*)carve(8000ull * 768 * 2);        // prot_data bf16 [8000x768]
    __bf16* WprotT   = (__bf16*)carve(256ull * 768 * 2);         // [256x768]
    __bf16* WjpT     = (__bf16*)carve(256ull * 256 * 2);         // [256x256]
    __bf16* fusedB   = (__bf16*)carve(1024ull * 1024 * 2);       // [1024x1024], valid 999x1000
    float*  prot     = (float*) carve(8000ull * 256 * 4);
    __bf16* reluprot = (__bf16*)carve(8000ull * 256 * 2);
    __bf16* qp_bf    = (__bf16*)carve((8000ull + 32) * 256 * 2); // +slack rows (M=1000 tiles/batch)
    float*  g1b      = (float*) carve(8ull * 56 * 256 * 4);
    float*  g2b      = (float*) carve(8ull * 56 * 256 * 4);
    float*  compb    = (float*) carve(8ull * 56 * 256 * 4);
    float*  qcf      = (float*) carve(8ull * 56 * 256 * 4);
    __bf16* qc_bf    = (__bf16*)carve((8ull * 56 + 16) * 256 * 2); // +slack rows (N=56 -> 64)
    float*  inter    = (float*) carve(8ull * 1000 * 56 * 4);     // logits->sigmoid->normalized
    __bf16* interT   = (__bf16*)carve(8ull * 64 * 1024 * 2);     // zero-filled, K-padded
    float*  fusedout = (float*) carve(8ull * 56 * 999 * 4);
    float*  accum    = (float*) carve(40ull * 4);                // zeroed scalars
    float*  sums     = accum;
    float*  l1a      = accum + 8;
    float*  d2a      = accum + 16;
    float*  rga      = accum + 24;
    float*  faa      = accum + 32;
    float*  cp_part  = (float*) carve(64ull * 2048 * 4);         // zeroed slices
    float*  sbuf     = (float*) carve(8000ull * 4);
    float*  cpb      = (float*) carve(2048ull * 4);
    float*  h2       = (float*) carve(8ull * 2048 * 4);
    float*  r1b      = (float*) carve(8ull * 600 * 4);
    float*  r2b      = (float*) carve(8ull * 300 * 4);

    const int T = 256;
    auto cdiv = [](int a, int b) { return (a + b - 1) / b; };

    k_zero<<<1, 64, 0, stream>>>(accum, 40);
    k_zero<<<cdiv(131072, T), T, 0, stream>>>(cp_part, 131072);          // 64*2048
    k_zero<<<cdiv(262144, T), T, 0, stream>>>((float*)interT, 262144);   // 8*64*1024 bf16
    k_f2bf<<<cdiv(6144000, T), T, 0, stream>>>(prot_data, A1, 6144000);
    k_transconv<<<cdiv(768 * 256, T), T, 0, stream>>>(W_prot, WprotT, 768, 256);
    k_transconv<<<cdiv(256 * 256, T), T, 0, stream>>>(Wjp, WjpT, 256, 256);
    k_f2bf_pad<<<cdiv(999000, T), T, 0, stream>>>(fused, fusedB, 999, 1000, 1024);

    // prot = prot_data @ W_prot + b_prot  (store f32 + relu'd bf16)
    wmma_gemm_bt<<<dim3(4, 125, 1), 128, 0, stream>>>(A1, WprotT, prot, reluprot, b_prot,
        8000, 256, 768, 768, 768, 256, 0, 0, 0, 1);
    // qp = relu(prot) @ Wjp + bjp  (bf16 only)
    wmma_gemm_bt<<<dim3(4, 125, 1), 128, 0, stream>>>(reluprot, WjpT, nullptr, qp_bf, bjp,
        8000, 256, 256, 256, 256, 256, 0, 0, 0, 0);

    // GCN stack (small dense layers)
    k_linear<<<448, 256, 0, stream>>>(drug_ver, gW0, gb0, g1b, nullptr, 43, 0);
    k_adjrelu<<<dim3(56, 8), 256, 0, stream>>>(drug_adj, g1b, g2b);
    k_linear<<<448, 256, 0, stream>>>(g2b, gW1, gb1, g1b, nullptr, 256, 0);
    k_adjrelu<<<dim3(56, 8), 256, 0, stream>>>(drug_adj, g1b, g2b);
    k_linear<<<448, 256, 0, stream>>>(g2b, gW2, gb2, g1b, nullptr, 256, 0);
    k_adjrelu<<<dim3(56, 8), 256, 0, stream>>>(drug_adj, g1b, g2b);
    k_linear<<<448, 256, 0, stream>>>(g2b, gWf, gbf, compb, nullptr, 256, 0);
    k_linear<<<448, 256, 0, stream>>>(compb, Wjc, bjc, qcf, qc_bf, 256, 1);

    // logits[b] = qp[b] @ qc[b]^T  (qc already [N,K] row-major)
    wmma_gemm_bt<<<dim3(1, 16, 8), 128, 0, stream>>>(qp_bf, qc_bf, inter, nullptr, nullptr,
        1000, 56, 256, 256, 256, 56, 256000, 14336, 56000, 0);

    k_sigsum<<<dim3(219, 8), 256, 0, stream>>>(inter, sums);
    k_norm<<<dim3(219, 8), 256, 0, stream>>>(inter, sums, interT, prot_int, exist, l1a, d2a);
    k_rowsq<<<cdiv(8000, T), T, 0, stream>>>(inter, sbuf);

    // fusedout[b] = interT[b] (56x1000, K-padded 1024) @ fused^T (fused already [N,K])
    wmma_gemm_bt<<<dim3(16, 1, 8), 128, 0, stream>>>(interT, fusedB, fusedout, nullptr, nullptr,
        56, 999, 1024, 1024, 1024, 999, 65536, 0, 55944, 0);
    k_fusedabs<<<dim3(219, 8), 256, 0, stream>>>(fusedout, faa);

    // fused tanh(prot * comp) * inter reduction (115M tanh, never materialized)
    k_cp<<<dim3(1000, 8), 256, 0, stream>>>(prot, compb, inter, cp_part);
    k_cpreduce<<<8, 256, 0, stream>>>(cp_part, cpb);
    // group regularizer (algebraically reduced to per-batch mat-vec)
    k_group<<<dim3(1000, 8), 256, 0, stream>>>(contacts, sbuf, rga);

    k_convpool<<<8, 256, 0, stream>>>(cpb, convw, convb, h2);
    k_mlp<<<dim3(3, 8), 256, 0, stream>>>(h2, Wr1, br1, r1b, 2048, 600);
    k_mlp<<<dim3(2, 8), 256, 0, stream>>>(r1b, Wr2, br2, r2b, 600, 300);
    k_final<<<1, 1, 0, stream>>>(l1a, faa, rga, d2a, r2b, Wr3, br3, label, (float*)d_out);
}
